// AttentionBlock_3058016715385
// MI455X (gfx1250) — compile-verified, matrix operand order checked by reading, not run on hardware
//
#include <hip/hip_runtime.h>

typedef __attribute__((ext_vector_type(16))) __bf16        v16bf;
typedef __attribute__((ext_vector_type(8)))  float         v8f;
typedef __attribute__((ext_vector_type(8)))  unsigned int  v8u;

__device__ __forceinline__ unsigned short f2bf(float f) {
    unsigned int u = __builtin_bit_cast(unsigned int, f);
    u += 0x7FFFu + ((u >> 16) & 1u);               // round-to-nearest-even
    return (unsigned short)(u >> 16);
}
__device__ __forceinline__ float bf2f(unsigned short s) {
    unsigned int u = ((unsigned int)s) << 16;
    return __builtin_bit_cast(float, u);
}
__device__ __forceinline__ v8f v8f_zero() {
    v8f z = {0.f, 0.f, 0.f, 0.f, 0.f, 0.f, 0.f, 0.f};
    return z;
}
__device__ __forceinline__ v8f wmma_bf16(v16bf a, v16bf b, v8f c) {
    // D = A(16x32 bf16) * B(32x16 bf16) + C(16x16 f32)
    return __builtin_amdgcn_wmma_f32_16x16x32_bf16(false, a, false, b, (short)0, c, false, false);
}

// CDNA5 16-bit fragment (16x32), k-contiguous storage: lane&15 = row (A) or col (B),
// half = lane>>4 selects the two 8-element k-runs.  Exactly two 16B loads.
__device__ __forceinline__ v16bf load_frag_k(const unsigned short* p, int row0, int k0, int pitch) {
    const int lane = threadIdx.x & 31;
    const int half = lane >> 4;
    const int m = row0 + (lane & 15);
    const unsigned short* base = p + (size_t)m * pitch + k0 + half * 8;
    uint4 a = *reinterpret_cast<const uint4*>(base);        // k = k0+half*8 .. +7
    uint4 b = *reinterpret_cast<const uint4*>(base + 16);   // k = k0+16+half*8 .. +7
    v8u u = {a.x, a.y, a.z, a.w, b.x, b.y, b.z, b.w};
    return __builtin_bit_cast(v16bf, u);
}

// ---------------------------------------------------------------- cast f32 -> bf16
__global__ __launch_bounds__(256) void cast_f32_bf16(const float* __restrict__ in,
                                                     unsigned short* __restrict__ out, int n) {
    int i = blockIdx.x * blockDim.x + threadIdx.x;
    int stride = gridDim.x * blockDim.x;
    for (; i < n; i += stride) out[i] = f2bf(in[i]);
}

// ---------------------------------------------------------------- bf16 GEMM, f32 accum
// C[M,N] = A[M,K] @ B[K,N]; A,B bf16 row-major.  Block tile 128x128, Ktile 64,
// 256 threads = 8 waves in a 2x4 grid, each wave owns 64x32 (4x2 WMMA tiles).
// B tile staged TRANSPOSED ([n][k]) so all fragment loads are 2x ds_load_b128.
template <bool OUT_F32>
__global__ __launch_bounds__(256) void gemm_bf16(const unsigned short* __restrict__ A,
                                                 const unsigned short* __restrict__ B,
                                                 void* __restrict__ Cp, int M, int N, int K) {
    constexpr int AP = 72;   // A tile pitch [m][k]: 64+8 bf16 -> 144B rows (16B aligned)
    constexpr int BPK = 72;  // B^T tile pitch [n][k]: 64+8 bf16
    __shared__ alignas(16) unsigned short As[128 * AP];
    __shared__ alignas(16) unsigned short Bs[128 * BPK];

    const int tid = threadIdx.x;
    const int wave = tid >> 5;
    const int wm = wave >> 2;  // 0..1
    const int wn = wave & 3;   // 0..3
    const int m0 = blockIdx.y * 128;
    const int n0 = blockIdx.x * 128;

    v8f acc[4][2];
#pragma unroll
    for (int i = 0; i < 4; ++i)
#pragma unroll
        for (int j = 0; j < 2; ++j) acc[i][j] = v8f_zero();

    for (int k0 = 0; k0 < K; k0 += 64) {
        // Stage A 128x64 (row-major, b128 copies).
#pragma unroll
        for (int i = 0; i < 4; ++i) {
            int idx = tid + i * 256;
            int ar = idx >> 3, ac = idx & 7;     // 128 rows x 8 uint4
            *reinterpret_cast<uint4*>(&As[ar * AP + ac * 8]) =
                *reinterpret_cast<const uint4*>(A + (size_t)(m0 + ar) * K + k0 + ac * 8);
        }
        // Stage B 64x128 transposed -> Bs[n][k], packing (k,k+1) pairs into dwords.
#pragma unroll
        for (int i = 0; i < 2; ++i) {
            int idx = tid + i * 256;             // 512 tasks: 32 k-pairs x 16 n-chunks
            int kp = idx & 31;                   // k-pair index
            int c = idx >> 5;                    // n-chunk of 8
            const unsigned short* g = B + (size_t)(k0 + 2 * kp) * N + n0 + c * 8;
            uint4 r0 = *reinterpret_cast<const uint4*>(g);
            uint4 r1 = *reinterpret_cast<const uint4*>(g + N);
            unsigned int w0[4] = {r0.x, r0.y, r0.z, r0.w};
            unsigned int w1[4] = {r1.x, r1.y, r1.z, r1.w};
            unsigned short* dst = &Bs[(c * 8) * BPK + 2 * kp];
#pragma unroll
            for (int j = 0; j < 4; ++j) {
                *reinterpret_cast<unsigned int*>(dst + (2 * j) * BPK) =
                    (w0[j] & 0xFFFFu) | (w1[j] << 16);
                *reinterpret_cast<unsigned int*>(dst + (2 * j + 1) * BPK) =
                    (w0[j] >> 16) | (w1[j] & 0xFFFF0000u);
            }
        }
        if (k0 + 64 < K) {   // gfx1250 global_prefetch_b8 hint for next K tile
            __builtin_prefetch(A + (size_t)(m0 + (tid >> 1)) * K + k0 + 64, 0, 0);
            __builtin_prefetch(B + (size_t)(k0 + 64 + (tid >> 2)) * N + n0, 0, 0);
        }
        __syncthreads();
#pragma unroll
        for (int kk = 0; kk < 64; kk += 32) {
            v16bf af[4], bf[2];
#pragma unroll
            for (int mt = 0; mt < 4; ++mt) af[mt] = load_frag_k(As, wm * 64 + mt * 16, kk, AP);
#pragma unroll
            for (int nt = 0; nt < 2; ++nt) bf[nt] = load_frag_k(Bs, wn * 32 + nt * 16, kk, BPK);
#pragma unroll
            for (int mt = 0; mt < 4; ++mt)
#pragma unroll
                for (int nt = 0; nt < 2; ++nt) acc[mt][nt] = wmma_bf16(af[mt], bf[nt], acc[mt][nt]);
        }
        __syncthreads();
    }

    const int lane = tid & 31;
    const int half = lane >> 4;
#pragma unroll
    for (int mt = 0; mt < 4; ++mt)
#pragma unroll
        for (int nt = 0; nt < 2; ++nt)
#pragma unroll
            for (int r = 0; r < 8; ++r) {
                int m = m0 + wm * 64 + mt * 16 + r + half * 8;  // C layout: m = r + half*8
                int n = n0 + wn * 32 + nt * 16 + (lane & 15);
                float val = acc[mt][nt][r];
                if constexpr (OUT_F32)
                    reinterpret_cast<float*>(Cp)[(size_t)m * N + n] = val;
                else
                    reinterpret_cast<unsigned short*>(Cp)[(size_t)m * N + n] = f2bf(val);
            }
}

// ---------------------------------------------------------------- RoPE + head repack
// qkv: bf16 [B*T, 6144]; outputs q/k/v bf16 [B*H*T, 128] (head-major).
__global__ __launch_bounds__(256) void rope_pack(const unsigned short* __restrict__ qkv,
                                                 unsigned short* __restrict__ qh,
                                                 unsigned short* __restrict__ kh,
                                                 unsigned short* __restrict__ vh) {
    int idx = blockIdx.x * blockDim.x + threadIdx.x;  // B*T*H*64 = 4,194,304 threads
    int d2 = idx & 63;
    int h = (idx >> 6) & 15;
    int t = (idx >> 10) & 2047;
    int b = idx >> 21;
    size_t row = ((size_t)(b * 2048 + t)) * 6144;
    int c = h * 128 + 2 * d2;
    float qe = bf2f(qkv[row + c]),        qo = bf2f(qkv[row + c + 1]);
    float ke = bf2f(qkv[row + 2048 + c]), ko = bf2f(qkv[row + 2048 + c + 1]);
    // inv_freq = theta^(-2*d2/128); log(10000) = 9.210340371976184
    float inv = __expf(-(float)(2 * d2) * (9.210340371976184f / 128.0f));
    float f = (float)t * inv;
    float sn, cs;
    __sincosf(f, &sn, &cs);
    size_t orow = ((size_t)((b * 16 + h) * 2048 + t)) * 128 + 2 * d2;
    qh[orow]     = f2bf(qe * cs - qo * sn);
    qh[orow + 1] = f2bf(qe * sn + qo * cs);
    kh[orow]     = f2bf(ke * cs - ko * sn);
    kh[orow + 1] = f2bf(ke * sn + ko * cs);
    vh[orow]     = qkv[row + 4096 + c];
    vh[orow + 1] = qkv[row + 4096 + c + 1];
}

// ---------------------------------------------------------------- flash attention
// One block per (qt, h, b): 4 waves, each owns 16 q rows of a 64-row q tile.
// S = Q K^T via WMMA over D, online softmax, O += P V via WMMA over Tk.
// V staged TRANSPOSED ([d][kr]) so P@V B-fragments are 2x ds_load_b128 as well.
__global__ __launch_bounds__(128) void flash_attn(const unsigned short* __restrict__ Q,
                                                  const unsigned short* __restrict__ Kh,
                                                  const unsigned short* __restrict__ Vh,
                                                  unsigned short* __restrict__ Y) {
    constexpr int KP = 136;  // K tile pitch [kr][d] (128+8)
    constexpr int VPK = 72;  // V^T tile pitch [d][kr] (64+8)
    constexpr int SP = 68;   // f32 score pitch
    constexpr int PP = 72;   // bf16 P pitch [m][kcol]
    __shared__ alignas(16) unsigned short Ks[64 * KP];
    __shared__ alignas(16) unsigned short Vs[128 * VPK];
    __shared__ float Sb[4][16 * SP];
    __shared__ alignas(16) unsigned short Pb[4][16 * PP];
    __shared__ float m_s[64], l_s[64], a_s[64];

    const int tid = threadIdx.x;
    const int lane = tid & 31;
    const int half = lane >> 4;
    const int wave = tid >> 5;
    const int qt = blockIdx.x;
    const int h = blockIdx.y;
    const int b = blockIdx.z;
    const int T = 2048, D = 128, E = 2048;
    const size_t headbase = ((size_t)(b * 16 + h)) * T;

    if (tid < 64) { m_s[tid] = -3.0e38f; l_s[tid] = 0.f; a_s[tid] = 0.f; }

    // Resident Q fragments: wave's 16 rows x 128 d, as 4 A-fragments (K=32 each).
    const unsigned short* qrow = Q + (headbase + (size_t)qt * 64 + wave * 16) * D;
    v16bf qf[4];
#pragma unroll
    for (int kd = 0; kd < 4; ++kd) qf[kd] = load_frag_k(qrow, 0, kd * 32, D);

    v8f acc[8];
#pragma unroll
    for (int dt = 0; dt < 8; ++dt) acc[dt] = v8f_zero();

    const float scale = 0.08838834764831845f;  // 1/sqrt(128)
    __syncthreads();

    for (int kt = 0; kt <= qt; ++kt) {
        // Stage K tile 64x128 row-major (b128 copies).
#pragma unroll
        for (int i = 0; i < 8; ++i) {
            int idx = tid + i * 128;
            int r = idx >> 4, c = idx & 15;
            *reinterpret_cast<uint4*>(&Ks[r * KP + c * 8]) =
                *reinterpret_cast<const uint4*>(Kh + (headbase + (size_t)kt * 64 + r) * D + c * 8);
        }
        // Stage V tile transposed -> Vs[d][kr], packing kr-pairs into dwords.
#pragma unroll
        for (int i = 0; i < 4; ++i) {
            int idx = tid + i * 128;             // 512 tasks: 32 kr-pairs x 16 d-chunks
            int kp = idx & 31;
            int c = idx >> 5;
            const unsigned short* g = Vh + (headbase + (size_t)kt * 64 + 2 * kp) * D + c * 8;
            uint4 r0 = *reinterpret_cast<const uint4*>(g);
            uint4 r1 = *reinterpret_cast<const uint4*>(g + D);
            unsigned int w0[4] = {r0.x, r0.y, r0.z, r0.w};
            unsigned int w1[4] = {r1.x, r1.y, r1.z, r1.w};
            unsigned short* dst = &Vs[(c * 8) * VPK + 2 * kp];
#pragma unroll
            for (int j = 0; j < 4; ++j) {
                *reinterpret_cast<unsigned int*>(dst + (2 * j) * VPK) =
                    (w0[j] & 0xFFFFu) | (w1[j] << 16);
                *reinterpret_cast<unsigned int*>(dst + (2 * j + 1) * VPK) =
                    (w0[j] >> 16) | (w1[j] & 0xFFFF0000u);
            }
        }
        __syncthreads();

        // Scores: 4 tiles of 16x16, contraction over D in 4 WMMA steps.
#pragma unroll
        for (int nt = 0; nt < 4; ++nt) {
            v8f s = v8f_zero();
#pragma unroll
            for (int kd = 0; kd < 4; ++kd)
                s = wmma_bf16(qf[kd], load_frag_k(Ks, nt * 16, kd * 32, KP), s);
            int nl = nt * 16 + (lane & 15);
            int kg = kt * 64 + nl;
#pragma unroll
            for (int r = 0; r < 8; ++r) {
                int ml = r + half * 8;
                int qg = qt * 64 + wave * 16 + ml;
                float v = s[r] * scale;
                Sb[wave][ml * SP + nl] = (kg > qg) ? -3.0e38f : v;  // causal mask
            }
        }
        __syncthreads();

        // Online softmax: 2 lanes per row (each handles 32 of the 64 columns).
        {
            int row = lane >> 1, seg = lane & 1;
            float* srow = &Sb[wave][row * SP + seg * 32];
            float mx = -3.0e38f;
#pragma unroll
            for (int j = 0; j < 32; ++j) mx = fmaxf(mx, srow[j]);
            mx = fmaxf(mx, __shfl_xor(mx, 1, 32));
            int gr = wave * 16 + row;
            float mold = m_s[gr];
            float mnew = fmaxf(mold, mx);
            float alpha = __expf(mold - mnew);
            float sum = 0.f;
            unsigned short* prow = &Pb[wave][row * PP + seg * 32];
#pragma unroll
            for (int j = 0; j < 32; ++j) {
                float p = __expf(srow[j] - mnew);
                sum += p;
                prow[j] = f2bf(p);
            }
            sum += __shfl_xor(sum, 1, 32);
            if (seg == 0) {
                m_s[gr] = mnew;
                l_s[gr] = alpha * l_s[gr] + sum;
                a_s[gr] = alpha;
            }
        }
        __syncthreads();

        // Rescale accumulators, then O += P @ V (contraction over 64 k rows, 2 chunks).
        float al[8];
#pragma unroll
        for (int r = 0; r < 8; ++r) al[r] = a_s[wave * 16 + r + half * 8];
#pragma unroll
        for (int dt = 0; dt < 8; ++dt)
#pragma unroll
            for (int r = 0; r < 8; ++r) acc[dt][r] *= al[r];
        v16bf pf[2];
#pragma unroll
        for (int kc = 0; kc < 2; ++kc) pf[kc] = load_frag_k(Pb[wave], 0, kc * 32, PP);
#pragma unroll
        for (int dt = 0; dt < 8; ++dt)
#pragma unroll
            for (int kc = 0; kc < 2; ++kc)
                acc[dt] = wmma_bf16(pf[kc], load_frag_k(Vs, dt * 16, kc * 32, VPK), acc[dt]);
        __syncthreads();
    }

    // Normalize and scatter to y[b*T + t, h*128 + d] (bf16).
    float rl[8];
#pragma unroll
    for (int r = 0; r < 8; ++r) rl[r] = 1.0f / l_s[wave * 16 + r + half * 8];
#pragma unroll
    for (int dt = 0; dt < 8; ++dt)
#pragma unroll
        for (int r = 0; r < 8; ++r) {
            int t = qt * 64 + wave * 16 + r + half * 8;
            int col = h * 128 + dt * 16 + (lane & 15);
            Y[((size_t)(b * T + t)) * E + col] = f2bf(acc[dt][r] * rl[r]);
        }
}

// ---------------------------------------------------------------- host launch
extern "C" void kernel_launch(void* const* d_in, const int* in_sizes, int n_in,
                              void* d_out, int out_size, void* d_ws, size_t ws_size,
                              hipStream_t stream) {
    (void)in_sizes; (void)n_in; (void)out_size; (void)ws_size;
    const float* x = (const float*)d_in[0];
    const float* Wqkv = (const float*)d_in[1];
    const float* Wout = (const float*)d_in[2];
    float* out = (float*)d_out;
    char* ws = (char*)d_ws;

    const size_t BT = 4096, E = 2048, E3 = 6144;
    // Workspace layout (all bf16 = ushort), ~160 MB total.
    unsigned short* xb    = (unsigned short*)(ws);
    unsigned short* wqkvb = (unsigned short*)(ws + BT * E * 2);
    unsigned short* woutb = (unsigned short*)(ws + BT * E * 2 + E * E3 * 2);
    unsigned short* qkvb  = (unsigned short*)(ws + BT * E * 2 + E * E3 * 2 + E * E * 2);
    unsigned short* qh = qkvb + BT * E3;
    unsigned short* kh = qh + BT * E;   // B*H*T*D == BT*E elements
    unsigned short* vh = kh + BT * E;
    unsigned short* yb = vh + BT * E;

    cast_f32_bf16<<<32768, 256, 0, stream>>>(x, xb, (int)(BT * E));
    cast_f32_bf16<<<49152, 256, 0, stream>>>(Wqkv, wqkvb, (int)(E * E3));
    cast_f32_bf16<<<16384, 256, 0, stream>>>(Wout, woutb, (int)(E * E));

    // qkv = x @ W_qkv   (4096 x 6144 x 2048)
    gemm_bf16<false><<<dim3(48, 32), 256, 0, stream>>>(xb, wqkvb, qkvb, 4096, 6144, 2048);
    // RoPE + repack to head-major q/k/v
    rope_pack<<<16384, 256, 0, stream>>>(qkvb, qh, kh, vh);
    // causal flash attention -> y (bf16, [B*T, E])
    flash_attn<<<dim3(32, 16, 2), 128, 0, stream>>>(qh, kh, vh, yb);
    // out = y @ W_out   (4096 x 2048 x 2048), f32 output
    gemm_bf16<true><<<dim3(16, 32), 256, 0, stream>>>(yb, woutb, out, 4096, 2048, 2048);
}
